// Net_75995151335950
// MI455X (gfx1250) — matrix operand order repaired and checked by reading, compiled, not run on hardware
//
#include <hip/hip_runtime.h>

// ---------------- problem constants ----------------
#define N_NODES 40000
#define E_EDGES 400000
#define ETOT    (E_EDGES + N_NODES)   // with self loops
#define HEADS   4
#define M_PAD   40064                 // 40000 rounded up to 128

typedef unsigned short u16t;
typedef unsigned int   u32t;

typedef __attribute__((ext_vector_type(16))) __bf16 v16bf;
typedef __attribute__((ext_vector_type(8)))  float  v8f;
typedef __attribute__((ext_vector_type(4)))  int    v4i;

// ---------------- fp32 -> bf16 pad/convert (row-major) ----------------
__global__ void cvt_pad_bf16(const float* __restrict__ src, u16t* __restrict__ dst,
                             int rows_src, int cols_src, int rows_dst, int cols_dst) {
    long long i = (long long)blockIdx.x * blockDim.x + threadIdx.x;
    long long total = (long long)rows_dst * cols_dst;
    if (i >= total) return;
    long long r = i / cols_dst;
    int c = (int)(i - r * cols_dst);
    float f = (r < rows_src && c < cols_src) ? src[r * (long long)cols_src + c] : 0.0f;
    u32t u = __float_as_uint(f);
    u += 0x7FFFu + ((u >> 16) & 1u);      // round-to-nearest-even
    dst[i] = (u16t)(u >> 16);
}

// fp32 [Wrows, Wcols] -> bf16 TRANSPOSED + padded [rows_dst=NcPad][cols_dst=Kpad]
__global__ void cvt_pad_bf16_T(const float* __restrict__ src, u16t* __restrict__ dst,
                               int rows_src, int cols_src, int rows_dst, int cols_dst) {
    long long i = (long long)blockIdx.x * blockDim.x + threadIdx.x;
    long long total = (long long)rows_dst * cols_dst;
    if (i >= total) return;
    long long r = i / cols_dst;               // n (output row = W column)
    int c = (int)(i - r * cols_dst);          // k (output col = W row)
    float f = (c < rows_src && r < cols_src) ? src[(size_t)c * cols_src + r] : 0.0f;
    u32t u = __float_as_uint(f);
    u += 0x7FFFu + ((u >> 16) & 1u);
    dst[i] = (u16t)(u >> 16);
}

// ---------------- WMMA bf16 GEMM: C[M_PAD,Nc] = A[M_PAD,K] x Bt[Nc,K]^T ----------------
// grid = (Nc/128, M_PAD/128), block = 256 threads (8 wave32 waves).
// Double-buffered LDS filled by GLOBAL_LOAD_ASYNC_TO_LDS_B128 (ASYNCcnt),
// each wave computes a 32x64 tile via 2x4 v_wmma_f32_16x16x32_bf16 per K-step.
#define LDSS 36   // padded row stride (bf16 elements), even -> u32-aligned pair reads

union AFrag { u32t u[8]; v16bf v; };

typedef __attribute__((address_space(1))) v4i* g4p;   // global int4*
typedef __attribute__((address_space(3))) v4i* l4p;   // LDS int4*

__device__ __forceinline__ void async_b128(const u16t* g, u16t* l) {
    __builtin_amdgcn_global_load_async_to_lds_b128((g4p)(void*)g, (l4p)(void*)l, 0, 0);
}

__global__ __launch_bounds__(256)
void gemm_bf16_wmma(const u16t* __restrict__ A, const u16t* __restrict__ Bt,
                    float* __restrict__ C, int Nc, int K) {
    __shared__ u16t As[2][128 * LDSS];
    __shared__ u16t Bs[2][128 * LDSS];

    const int tid  = threadIdx.x;
    const int lane = tid & 31;
    const int wave = tid >> 5;
    const int half = lane >> 4;     // 0 / 1
    const int m16  = lane & 15;
    const int wr   = wave >> 1;     // 0..3 -> 32-row slab
    const int wc   = wave & 1;      // 0..1 -> 64-col slab
    const int blockM = blockIdx.y * 128;
    const int blockN = blockIdx.x * 128;

    // this thread's two 16B chunks of each 128x32 tile: li in {tid, tid+256} of 512
    const int row0 = tid >> 2,         g0 = tid & 3;          // chunk 0
    const int row1 = (tid + 256) >> 2, g1 = (tid + 256) & 3;  // chunk 1

    v8f acc[2][4];
    #pragma unroll
    for (int i = 0; i < 2; i++)
        #pragma unroll
        for (int j = 0; j < 4; j++)
            acc[i][j] = (v8f){0.f,0.f,0.f,0.f,0.f,0.f,0.f,0.f};

    // ---- async tile fetch: A rows [blockM,+128), Bt rows [blockN,+128), k-slice [k0,+32) ----
    auto issue_tile = [&](int k0, int p) {
        async_b128(A  + (size_t)(blockM + row0) * K + k0 + g0 * 8, &As[p][row0 * LDSS + g0 * 8]);
        async_b128(A  + (size_t)(blockM + row1) * K + k0 + g1 * 8, &As[p][row1 * LDSS + g1 * 8]);
        async_b128(Bt + (size_t)(blockN + row0) * K + k0 + g0 * 8, &Bs[p][row0 * LDSS + g0 * 8]);
        async_b128(Bt + (size_t)(blockN + row1) * K + k0 + g1 * 8, &Bs[p][row1 * LDSS + g1 * 8]);
    };

    int p = 0;
    issue_tile(0, 0);                                   // prologue: K-slice 0 -> buf 0

    for (int k0 = 0; k0 < K; k0 += 32) {
        const bool more = (k0 + 32) < K;
        __syncthreads();                                // readers of buf p^1 are done
        if (more) issue_tile(k0 + 32, p ^ 1);           // prefetch next slice
        if (more) __builtin_amdgcn_s_wait_asynccnt(4);  // retire this slice's 4 older ops
        else      __builtin_amdgcn_s_wait_asynccnt(0);
        __syncthreads();                                // buf p fully resident

        // ---- build fragments per documented CDNA5 layouts ----
        AFrag af[2], bf[4];
        #pragma unroll
        for (int mt = 0; mt < 2; mt++) {
            const u16t* base = &As[p][(wr * 32 + mt * 16 + m16) * LDSS];
            #pragma unroll
            for (int i = 0; i < 8; i++) {
                int k = ((i < 4) ? 2 * i : 16 + 2 * (i - 4)) + 8 * half;  // A: interleaved halves
                af[mt].u[i] = *(const u32t*)(base + k);
            }
        }
        #pragma unroll
        for (int nt = 0; nt < 4; nt++) {
            const u16t* base = &Bs[p][(wc * 64 + nt * 16 + m16) * LDSS];
            #pragma unroll
            for (int i = 0; i < 8; i++) {
                int k = ((i < 4) ? 2 * i : 16 + 2 * (i - 4)) + 8 * half;  // B: interleaved halves
                bf[nt].u[i] = *(const u32t*)(base + k);
            }
        }
        #pragma unroll
        for (int mt = 0; mt < 2; mt++)
            #pragma unroll
            for (int nt = 0; nt < 4; nt++)
                acc[mt][nt] = __builtin_amdgcn_wmma_f32_16x16x32_bf16(
                    false, af[mt].v, false, bf[nt].v, (short)0, acc[mt][nt], false, false);
        p ^= 1;
    }

    // ---- store: C/D layout: vgpr i -> M = i + 8*half, N = m16 ----
    #pragma unroll
    for (int mt = 0; mt < 2; mt++)
        #pragma unroll
        for (int nt = 0; nt < 4; nt++) {
            int col   = blockN + wc * 64 + nt * 16 + m16;
            int rbase = blockM + wr * 32 + mt * 16 + 8 * half;
            #pragma unroll
            for (int i = 0; i < 8; i++)
                C[(size_t)(rbase + i) * Nc + col] = acc[mt][nt][i];
        }
}

// ---------------- attention helpers ----------------
__device__ __forceinline__ u32t fenc(float f) {
    u32t u = __float_as_uint(f);
    return (u & 0x80000000u) ? ~u : (u | 0x80000000u);
}
__device__ __forceinline__ float fdec(u32t e) {
    u32t u = (e & 0x80000000u) ? (e & 0x7FFFFFFFu) : ~e;
    return __uint_as_float(u);
}

// per (node, head): dot(feat[n,h,:], a_src/a_dst[h,:])
__global__ void node_logits(const float* __restrict__ feat, int ldF,
                            const float* __restrict__ a_s, const float* __restrict__ a_d,
                            float* __restrict__ als, float* __restrict__ ald, int C) {
    int i = blockIdx.x * blockDim.x + threadIdx.x;
    if (i >= N_NODES * HEADS) return;
    int n = i >> 2, h = i & 3;
    const float* fp = feat + (size_t)n * ldF + h * C;
    float ss = 0.f, sd = 0.f;
    for (int c = 0; c < C; c++) { float f = fp[c]; ss += f * a_s[h * C + c]; sd += f * a_d[h * C + c]; }
    als[i] = ss; ald[i] = sd;
}

__global__ void init_nodes(u32t* __restrict__ menc, float* __restrict__ denom) {
    int i = blockIdx.x * blockDim.x + threadIdx.x;
    if (i >= N_NODES * HEADS) return;
    menc[i] = fenc(-3.402823466e38f);
    denom[i] = 0.f;
}

__global__ void edge_logits(const int* __restrict__ ei,
                            const float* __restrict__ als, const float* __restrict__ ald,
                            float* __restrict__ ebuf, u32t* __restrict__ menc) {
    int e = blockIdx.x * blockDim.x + threadIdx.x;
    if (e >= ETOT) return;
    int s = (e < E_EDGES) ? ei[e] : (e - E_EDGES);
    int d = (e < E_EDGES) ? ei[E_EDGES + e] : (e - E_EDGES);
    #pragma unroll
    for (int h = 0; h < HEADS; h++) {
        float l = als[s * HEADS + h] + ald[d * HEADS + h];
        l = (l > 0.f) ? l : 0.2f * l;                  // leaky_relu(0.2)
        ebuf[(size_t)e * HEADS + h] = l;
        atomicMax(&menc[d * HEADS + h], fenc(l));
    }
}

__global__ void edge_exp(const int* __restrict__ ei, float* __restrict__ ebuf,
                         const u32t* __restrict__ menc, float* __restrict__ denom) {
    long long i = (long long)blockIdx.x * blockDim.x + threadIdx.x;
    if (i >= (long long)ETOT * HEADS) return;
    int e = (int)(i >> 2), h = (int)(i & 3);
    int d = (e < E_EDGES) ? ei[E_EDGES + e] : (e - E_EDGES);
    float ex = __expf(ebuf[i] - fdec(menc[d * HEADS + h]));
    ebuf[i] = ex;
    atomicAdd(&denom[d * HEADS + h], ex);
}

__global__ void edge_norm(const int* __restrict__ ei, float* __restrict__ ebuf,
                          const float* __restrict__ denom) {
    long long i = (long long)blockIdx.x * blockDim.x + threadIdx.x;
    if (i >= (long long)ETOT * HEADS) return;
    int e = (int)(i >> 2), h = (int)(i & 3);
    int d = (e < E_EDGES) ? ei[E_EDGES + e] : (e - E_EDGES);
    ebuf[i] = ebuf[i] / (denom[d * HEADS + h] + 1e-16f);
}

// scatter-add messages: agg[dst, h*C+c] += w[e,h] * feat[src, h*C+c]
__global__ void scatter_msgs(const int* __restrict__ ei, const float* __restrict__ feat, int ldF,
                             const float* __restrict__ ebuf, float* __restrict__ agg, int C) {
    long long t = (long long)blockIdx.x * blockDim.x + threadIdx.x;
    int hc_span = HEADS * C;
    long long total = (long long)ETOT * hc_span;
    if (t >= total) return;
    long long e = t / hc_span;
    int hc = (int)(t - e * hc_span);
    int h = hc / C;
    int s = (e < E_EDGES) ? ei[e] : (int)(e - E_EDGES);
    int d = (e < E_EDGES) ? ei[E_EDGES + e] : (int)(e - E_EDGES);
    float w = ebuf[e * HEADS + h];
    atomicAdd(&agg[(size_t)d * hc_span + hc], w * feat[(size_t)s * ldF + hc]);
}

// layers 1,2: out = relu(agg + res + bias), cols = H*C = ld of res
__global__ void combine_relu(const float* __restrict__ agg, const float* __restrict__ res,
                             const float* __restrict__ bias, float* __restrict__ out, int cols) {
    long long t = (long long)blockIdx.x * blockDim.x + threadIdx.x;
    if (t >= (long long)N_NODES * cols) return;
    int c = (int)(t % cols);
    float v = agg[t] + res[t] + bias[c];
    out[t] = (v > 0.f) ? v : 0.f;
}

// layer 3: out[n,c] = mean_h agg[n, h*121+c] + res[n*128+c] + bias[c]
__global__ void combine_mean(const float* __restrict__ agg, const float* __restrict__ res,
                             const float* __restrict__ bias, float* __restrict__ out) {
    long long t = (long long)blockIdx.x * blockDim.x + threadIdx.x;
    if (t >= (long long)N_NODES * 121) return;
    int n = (int)(t / 121), c = (int)(t % 121);
    float s = 0.f;
    #pragma unroll
    for (int h = 0; h < HEADS; h++) s += agg[(size_t)n * 484 + h * 121 + c];
    out[t] = 0.25f * s + res[(size_t)n * 128 + c] + bias[c];
}

// ---------------- host-side layer driver ----------------
static void run_layer(const u16t* Ab, int K,
                      const float* W,  int Wrows, int Wcols, int NcPad,
                      const float* Wr, int WrCols, int NcRPad,
                      const float* a_s, const float* a_d, const float* bias, int C,
                      u16t* Wb, u16t* Wrb, float* feat, float* res, float* agg,
                      float* ebuf, float* als, float* ald, u32t* menc, float* denom,
                      const int* eidx, float* outbuf, int is_layer3, hipStream_t stream) {
    const int B = 256;
    long long t;

    // weights: transpose+pad to [NcPad][K] bf16 so GEMM B-tiles fetch like A-tiles
    t = (long long)NcPad * K;
    cvt_pad_bf16_T<<<(unsigned)((t + B - 1) / B), B, 0, stream>>>(W, Wb, Wrows, Wcols, NcPad, K);
    t = (long long)NcRPad * K;
    cvt_pad_bf16_T<<<(unsigned)((t + B - 1) / B), B, 0, stream>>>(Wr, Wrb, Wrows, WrCols, NcRPad, K);

    gemm_bf16_wmma<<<dim3(NcPad / 128, M_PAD / 128), B, 0, stream>>>(Ab, Wb, feat, NcPad, K);
    gemm_bf16_wmma<<<dim3(NcRPad / 128, M_PAD / 128), B, 0, stream>>>(Ab, Wrb, res, NcRPad, K);

    t = (long long)N_NODES * HEADS;
    node_logits<<<(unsigned)((t + B - 1) / B), B, 0, stream>>>(feat, NcPad, a_s, a_d, als, ald, C);
    init_nodes<<<(unsigned)((t + B - 1) / B), B, 0, stream>>>(menc, denom);

    edge_logits<<<(ETOT + B - 1) / B, B, 0, stream>>>(eidx, als, ald, ebuf, menc);
    t = (long long)ETOT * HEADS;
    edge_exp<<<(unsigned)((t + B - 1) / B), B, 0, stream>>>(eidx, ebuf, menc, denom);
    edge_norm<<<(unsigned)((t + B - 1) / B), B, 0, stream>>>(eidx, ebuf, denom);

    (void)hipMemsetAsync(agg, 0, (size_t)N_NODES * HEADS * C * sizeof(float), stream);
    t = (long long)ETOT * HEADS * C;
    scatter_msgs<<<(unsigned)((t + B - 1) / B), B, 0, stream>>>(eidx, feat, NcPad, ebuf, agg, C);

    if (!is_layer3) {
        t = (long long)N_NODES * HEADS * C;
        combine_relu<<<(unsigned)((t + B - 1) / B), B, 0, stream>>>(agg, res, bias, outbuf, HEADS * C);
    } else {
        t = (long long)N_NODES * 121;
        combine_mean<<<(unsigned)((t + B - 1) / B), B, 0, stream>>>(agg, res, bias, outbuf);
    }
}

// ---------------- entry point ----------------
extern "C" void kernel_launch(void* const* d_in, const int* in_sizes, int n_in,
                              void* d_out, int out_size, void* d_ws, size_t ws_size,
                              hipStream_t stream) {
    const float* x    = (const float*)d_in[0];
    const int*   eidx = (const int*)d_in[1];
    const float* W1  = (const float*)d_in[2];
    const float* a1s = (const float*)d_in[3];
    const float* a1d = (const float*)d_in[4];
    const float* Wr1 = (const float*)d_in[5];
    const float* b1  = (const float*)d_in[6];
    const float* W2  = (const float*)d_in[7];
    const float* a2s = (const float*)d_in[8];
    const float* a2d = (const float*)d_in[9];
    const float* Wr2 = (const float*)d_in[10];
    const float* b2  = (const float*)d_in[11];
    const float* W3  = (const float*)d_in[12];
    const float* a3s = (const float*)d_in[13];
    const float* a3d = (const float*)d_in[14];
    const float* Wr3 = (const float*)d_in[15];
    const float* b3  = (const float*)d_in[16];
    float* out = (float*)d_out;

    // ---- workspace carve-out (256B aligned slices) ----
    char* base = (char*)d_ws;
    size_t off = 0;
    auto carve = [&](size_t bytes) { void* p = base + off; off += (bytes + 255) & ~(size_t)255; return p; };

    u16t* Ab    = (u16t*)carve((size_t)M_PAD * 256 * 2);     // bf16 GEMM A (max K=256)
    u16t* Wb    = (u16t*)carve((size_t)512 * 256 * 2);       // bf16 W^T (max 512x256)
    u16t* Wrb   = (u16t*)carve((size_t)256 * 256 * 2);       // bf16 Wres^T
    float* feat = (float*)carve((size_t)M_PAD * 512 * 4);    // GEMM feat out (max Nc=512)
    float* res  = (float*)carve((size_t)M_PAD * 256 * 4);    // GEMM residual out
    float* hbuf = (float*)carve((size_t)N_NODES * 256 * 4);  // layer activations fp32
    float* agg  = (float*)carve((size_t)N_NODES * 484 * 4);  // message aggregation (max layer3)
    float* ebuf = (float*)carve((size_t)ETOT * HEADS * 4);   // per-edge attention
    float* als  = (float*)carve((size_t)N_NODES * HEADS * 4);
    float* ald  = (float*)carve((size_t)N_NODES * HEADS * 4);
    u32t* menc  = (u32t*)carve((size_t)N_NODES * HEADS * 4);
    float* denom= (float*)carve((size_t)N_NODES * HEADS * 4);

    const int B = 256;
    long long t;

    // ---- layer 1: x [N,50] -> pad to bf16 [M_PAD,64] ----
    t = (long long)M_PAD * 64;
    cvt_pad_bf16<<<(unsigned)((t + B - 1) / B), B, 0, stream>>>(x, Ab, N_NODES, 50, M_PAD, 64);
    run_layer(Ab, 64, W1, 50, 256, 256, Wr1, 256, 256, a1s, a1d, b1, 64,
              Wb, Wrb, feat, res, agg, ebuf, als, ald, menc, denom, eidx,
              hbuf, 0, stream);

    // ---- layer 2: h1 [N,256] -> bf16 [M_PAD,256] ----
    t = (long long)M_PAD * 256;
    cvt_pad_bf16<<<(unsigned)((t + B - 1) / B), B, 0, stream>>>(hbuf, Ab, N_NODES, 256, M_PAD, 256);
    run_layer(Ab, 256, W2, 256, 256, 256, Wr2, 256, 256, a2s, a2d, b2, 64,
              Wb, Wrb, feat, res, agg, ebuf, als, ald, menc, denom, eidx,
              hbuf, 0, stream);

    // ---- layer 3: h2 [N,256] -> bf16 [M_PAD,256]; W3 padded to 512 cols, Wr3 to 128 ----
    t = (long long)M_PAD * 256;
    cvt_pad_bf16<<<(unsigned)((t + B - 1) / B), B, 0, stream>>>(hbuf, Ab, N_NODES, 256, M_PAD, 256);
    run_layer(Ab, 256, W3, 256, 484, 512, Wr3, 121, 128, a3s, a3d, b3, 121,
              Wb, Wrb, feat, res, agg, ebuf, als, ald, menc, denom, eidx,
              out, 1, stream);
}
